// SignalVectorQuantizedVAE_11287174053932
// MI455X (gfx1250) — matrix operand order repaired and checked by reading, not compiled, not run on hardware
//
#include <hip/hip_runtime.h>

#define B_   32
#define H_   64
#define D_   256
#define C_   16
#define K_   2048
#define T_   15
#define PAD_ 7
#define M_   (B_*C_*H_)   // 32768 rows for the VQ GEMM
#define ROWS_ (B_*H_)     // 2048 (b,h) rows
#define EPS_ 1e-5f

typedef __attribute__((ext_vector_type(16))) __bf16 v16bf;
typedef __attribute__((ext_vector_type(8)))  float  v8f;

static __device__ __forceinline__ unsigned short f2bf(float f) {
    unsigned u = __float_as_uint(f);
    u += 0x7fffu + ((u >> 16) & 1u);   // round-to-nearest-even
    return (unsigned short)(u >> 16);
}

// ---------------------------------------------------------------------------
// Pass 1: per-block partial sums of conv output (sum, sumsq) per channel.
// Deterministic: fixed 256 blocks, tree reduction, no float atomics.
// ---------------------------------------------------------------------------
__global__ void k_stats_partial(const float* __restrict__ x,
                                const float* __restrict__ w,
                                float* __restrict__ part) {
    __shared__ float xs[D_];
    __shared__ float wl[C_*T_];
    __shared__ float red[256];
    const int tid = threadIdx.x;
    if (tid < C_*T_) wl[tid] = w[tid];
    float aS[C_], aQ[C_];
    #pragma unroll
    for (int c = 0; c < C_; ++c) { aS[c] = 0.f; aQ[c] = 0.f; }
    const int row0 = blockIdx.x * 8;                // 2048 rows / 256 blocks
    for (int j = 0; j < 8; ++j) {
        __syncthreads();
        xs[tid] = x[(row0 + j) * D_ + tid];
        __syncthreads();
        #pragma unroll
        for (int c = 0; c < C_; ++c) {
            float v = 0.f;
            #pragma unroll
            for (int t = 0; t < T_; ++t) {
                int idx = tid + t - PAD_;
                float xv = (idx >= 0 && idx < D_) ? xs[idx] : 0.f;
                v = fmaf(wl[c*T_ + t], xv, v);
            }
            aS[c] += v;
            aQ[c] = fmaf(v, v, aQ[c]);
        }
    }
    for (int q = 0; q < 32; ++q) {
        float v = (q < 16) ? aS[q] : aQ[q - 16];
        red[tid] = v; __syncthreads();
        for (int s = 128; s > 0; s >>= 1) {
            if (tid < s) red[tid] += red[tid + s];
            __syncthreads();
        }
        if (tid == 0) part[blockIdx.x * 32 + q] = red[0];
        __syncthreads();
    }
}

// ---------------------------------------------------------------------------
// Pass 2: fold partials -> per-channel scale/bias for BatchNorm (train mode).
// ---------------------------------------------------------------------------
__global__ void k_stats_final(const float* __restrict__ part,
                              const float* __restrict__ gamma,
                              const float* __restrict__ beta,
                              float* __restrict__ sb) {
    const int j = threadIdx.x;                      // 32 threads
    float s = 0.f;
    for (int b = 0; b < 256; ++b) s += part[b * 32 + j];
    __shared__ float tot[32];
    tot[j] = s; __syncthreads();
    if (j < C_) {
        const float N = (float)(B_ * H_ * D_);
        float mean = tot[j] / N;
        float var  = tot[16 + j] / N - mean * mean;
        float sc   = gamma[j] * rsqrtf(var + EPS_);
        sb[j]      = sc;
        sb[16 + j] = beta[j] - mean * sc;
    }
}

// ---------------------------------------------------------------------------
// Pass 3: recompute conv, apply BN -> z_e (f32 to d_out, bf16 to workspace).
// ---------------------------------------------------------------------------
__global__ void k_normalize(const float* __restrict__ x,
                            const float* __restrict__ w,
                            const float* __restrict__ sb,
                            float* __restrict__ ze,
                            unsigned short* __restrict__ zbf) {
    __shared__ float xs[D_];
    __shared__ float wl[C_*T_];
    __shared__ float sbl[32];
    const int tid = threadIdx.x;
    const int row = blockIdx.x;                     // b*H + h
    const int b = row / H_, h = row % H_;
    if (tid < C_*T_) wl[tid] = w[tid];
    if (tid < 32)    sbl[tid] = sb[tid];
    xs[tid] = x[row * D_ + tid];
    __syncthreads();
    #pragma unroll
    for (int c = 0; c < C_; ++c) {
        float v = 0.f;
        #pragma unroll
        for (int t = 0; t < T_; ++t) {
            int idx = tid + t - PAD_;
            float xv = (idx >= 0 && idx < D_) ? xs[idx] : 0.f;
            v = fmaf(wl[c*T_ + t], xv, v);
        }
        v = fmaf(sbl[c], v, sbl[16 + c]);
        const int m = (b * C_ + c) * H_ + h;
        ze [m * D_ + tid] = v;
        zbf[m * D_ + tid] = f2bf(v);
    }
}

// ---------------------------------------------------------------------------
// Pass 4: codebook -> bf16 + en[k] = 0.5*||e_k||^2
// ---------------------------------------------------------------------------
__global__ void k_embprep(const float* __restrict__ emb,
                          unsigned short* __restrict__ ebf,
                          float* __restrict__ en) {
    __shared__ float red[256];
    const int k = blockIdx.x, d = threadIdx.x;
    float v = emb[k * D_ + d];
    ebf[k * D_ + d] = f2bf(v);
    red[d] = v * v; __syncthreads();
    for (int s = 128; s > 0; s >>= 1) {
        if (d < s) red[d] += red[d + s];
        __syncthreads();
    }
    if (d == 0) en[k] = 0.5f * red[0];
}

// ---------------------------------------------------------------------------
// Pass 5: fused GEMM (32768x2048x256, bf16 WMMA, f32 acc) + row-argmin.
// Block = 128 threads = 4 waves; each wave owns a 16-row M tile (A in VGPRs).
// B tiles (32 codewords/iter, 16 KB) are DMA'd global->LDS once per block via
// global_load_async_to_lds_b128 with double buffering (ASYNCcnt + barrier),
// shared by all 4 waves. Two independent accumulator chains per iteration
// remove the WMMA->WMMA RAW hazard stalls.
// score[m,k] = 0.5||e_k||^2 - z_m . e_k  (argmin-equivalent to L2 distance)
// ---------------------------------------------------------------------------
__global__ void __launch_bounds__(128, 1)
k_vq_argmin(const unsigned short* __restrict__ zbf,
            const unsigned short* __restrict__ ebf,
            const float* __restrict__ en,
            int* __restrict__ latents) {
    __shared__ __align__(16) char smem[32768];      // A stage, then 2x16KB B bufs
    const int tid = threadIdx.x;
    const int m0  = blockIdx.x * 64;

    // ---- stage A: 64 z_e rows (32 KB), coalesced ----
    {
        const uint4* src = (const uint4*)(zbf + (size_t)m0 * D_);
        uint4* dst = (uint4*)smem;
        #pragma unroll
        for (int i = tid; i < 2048; i += 128) dst[i] = src[i];
    }
    __syncthreads();

    const int lane = tid & 31;
    const int wave = tid >> 5;
    const int l    = lane & 15;
    const int half = lane >> 4;

    union Frag { v16bf v; uint4 u[2]; };

    // Preload A fragments (8 k-steps of 32) for this wave's 16 rows.
    // CDNA5 16-bit A 16x32: lane<16 holds K[0..7],K[16..23]; lane>=16 K[8..15],K[24..31]
    Frag a[8];
    {
        const uint4* ar = (const uint4*)smem + (wave * 16 + l) * (D_ / 8);
        #pragma unroll
        for (int kk = 0; kk < 8; ++kk) {
            a[kk].u[0] = ar[kk * 4 + half];
            a[kk].u[1] = ar[kk * 4 + half + 2];
        }
    }
    __syncthreads();   // A region now reusable as B double-buffer

    // Async DMA of one 32-codeword emb tile (16 KB) into LDS buffer `buf`.
    // LDS byte address = low 32 bits of the flat pointer (ISA flat->LDS trunc).
    auto issue_pair = [&](int p, int buf) {
        const char* g = (const char*)(ebf + (size_t)p * 32 * D_);
        unsigned lbase = (unsigned)(size_t)(smem + buf * 16384);
        #pragma unroll
        for (int j = 0; j < 8; ++j) {
            const int ci = tid + j * 128;           // 1024 chunks of 16 B
            unsigned laddr = lbase + ci * 16;
            const char* gaddr = g + (size_t)ci * 16;
            asm volatile("global_load_async_to_lds_b128 %0, %1, off"
                         :: "v"(laddr), "v"(gaddr) : "memory");
        }
    };

    float best[8]; int bestk[8];
    #pragma unroll
    for (int r = 0; r < 8; ++r) { best[r] = 3.4e38f; bestk[r] = 0; }

    issue_pair(0, 0);
    asm volatile("s_wait_asynccnt 0x0" ::: "memory");
    __syncthreads();

    for (int p = 0; p < K_ / 32; ++p) {             // 64 iterations
        if (p < 63) issue_pair(p + 1, (p + 1) & 1); // overlap DMA with compute
        const uint4* bt = (const uint4*)(smem + (p & 1) * 16384);
        // B 32x16 bf16 frag: lane<16 -> K[0..7],K[16..23]; lane>=16 -> K[8..15],K[24..31]
        const uint4* b0 = bt + (l)      * (D_ / 8); // tile 0: local rows 0..15
        const uint4* b1 = bt + (16 + l) * (D_ / 8); // tile 1: local rows 16..31
        v8f acc0 = {}, acc1 = {};
        #pragma unroll
        for (int kk = 0; kk < 8; ++kk) {
            Frag f0, f1;
            f0.u[0] = b0[kk * 4 + half]; f0.u[1] = b0[kk * 4 + half + 2];
            f1.u[0] = b1[kk * 4 + half]; f1.u[1] = b1[kk * 4 + half + 2];
            acc0 = __builtin_amdgcn_wmma_f32_16x16x32_bf16(
                false, a[kk].v, false, f0.v, (short)0, acc0, false, false);
            acc1 = __builtin_amdgcn_wmma_f32_16x16x32_bf16(
                false, a[kk].v, false, f1.v, (short)0, acc1, false, false);
        }
        const int n0 = p * 32;
        const int kc0 = n0 + l, kc1 = n0 + 16 + l;
        const float e0 = en[kc0], e1 = en[kc1];
        #pragma unroll
        for (int r = 0; r < 8; ++r) {
            float s0 = e0 - acc0[r];
            if (s0 < best[r]) { best[r] = s0; bestk[r] = kc0; }
            float s1 = e1 - acc1[r];
            if (s1 < best[r]) { best[r] = s1; bestk[r] = kc1; }
        }
        if (p < 63) {
            asm volatile("s_wait_asynccnt 0x0" ::: "memory");
            __syncthreads();
        }
    }

    // D layout: VGPR r -> row r (lanes 0-15) / row 8+r (lanes 16-31), col = lane&15.
    // Reduce min across the 16 lanes sharing a row (xor masks 1..8 stay in-half).
    #pragma unroll
    for (int r = 0; r < 8; ++r) {
        float bv = best[r]; int bk = bestk[r];
        #pragma unroll
        for (int off = 8; off >= 1; off >>= 1) {
            float ov = __shfl_xor(bv, off, 32);
            int   ok = __shfl_xor(bk, off, 32);
            if (ov < bv || (ov == bv && ok < bk)) { bv = ov; bk = ok; }
        }
        if (l == 0) latents[m0 + wave * 16 + half * 8 + r] = bk;
    }
}

// ---------------------------------------------------------------------------
// Pass 6: z_q gather (writes z_q output) + transposed-conv + tanh -> x_tilde.
// ---------------------------------------------------------------------------
__global__ void k_decode(const float* __restrict__ emb,
                         const int* __restrict__ latents,
                         const float* __restrict__ wdec,
                         float* __restrict__ zq,
                         float* __restrict__ xt) {
    __shared__ float zr[C_][D_];
    __shared__ float wl[C_*T_];
    const int tid = threadIdx.x;
    const int row = blockIdx.x;                     // b*H + h
    const int b = row / H_, h = row % H_;
    if (tid < C_*T_) wl[tid] = wdec[tid];
    #pragma unroll
    for (int c = 0; c < C_; ++c) {
        const int m = (b * C_ + c) * H_ + h;
        const int k = latents[m];
        float v = emb[(size_t)k * D_ + tid];
        zr[c][tid] = v;
        zq[(size_t)m * D_ + tid] = v;
    }
    __syncthreads();
    // transposed conv (stride 1) == conv with flipped kernel, pad k-1-p = 7:
    // out[d] = sum_c sum_s z_q[c][d + 7 - s] * w_dec[c][s]
    float s = 0.f;
    #pragma unroll
    for (int c = 0; c < C_; ++c) {
        #pragma unroll
        for (int t = 0; t < T_; ++t) {
            int idx = tid + PAD_ - t;
            float zv = (idx >= 0 && idx < D_) ? zr[c][idx] : 0.f;
            s = fmaf(wl[c*T_ + t], zv, s);
        }
    }
    xt[row * D_ + tid] = tanhf(s);
}

// ---------------------------------------------------------------------------
extern "C" void kernel_launch(void* const* d_in, const int* in_sizes, int n_in,
                              void* d_out, int out_size, void* d_ws, size_t ws_size,
                              hipStream_t stream) {
    (void)in_sizes; (void)n_in; (void)out_size; (void)ws_size;
    const float* x     = (const float*)d_in[0];   // (32,1,64,256)
    const float* w_enc = (const float*)d_in[1];   // (16,1,1,15)
    const float* gamma = (const float*)d_in[2];   // (16,)
    const float* beta  = (const float*)d_in[3];   // (16,)
    const float* emb   = (const float*)d_in[4];   // (2048,256)
    const float* w_dec = (const float*)d_in[5];   // (16,1,1,15)

    float* out = (float*)d_out;
    float* xt = out;                              // 524288  (x_tilde)
    float* ze = out + 524288;                     // 8388608 (z_e)
    float* zq = ze + 8388608;                     // 8388608 (z_q)

    char* ws = (char*)d_ws;
    unsigned short* ebf = (unsigned short*)ws;               // 1 MB  emb bf16
    float* en   = (float*)(ws + (1u << 20));                 // 8 KB  0.5*||e_k||^2
    float* sb   = en + K_;                                   // 128 B scale/bias
    float* part = sb + 32;                                   // 32 KB partial stats
    int*   lat  = (int*)(part + 256 * 32);                   // 128 KB latents
    unsigned short* zbf = (unsigned short*)(lat + M_);       // 16 MB z_e bf16

    k_stats_partial<<<256, 256, 0, stream>>>(x, w_enc, part);
    k_stats_final  <<<1,   32,  0, stream>>>(part, gamma, beta, sb);
    k_normalize    <<<ROWS_, 256, 0, stream>>>(x, w_enc, sb, ze, zbf);
    k_embprep      <<<K_,   256, 0, stream>>>(emb, ebf, en);
    k_vq_argmin    <<<M_ / 64, 128, 0, stream>>>(zbf, ebf, en, lat);
    k_decode       <<<ROWS_, 256, 0, stream>>>(emb, lat, w_dec, zq, xt);
}
